// MambaLayer4MTP_51539607552221
// MI455X (gfx1250) — matrix operand order repaired and checked by reading, not compiled, not run on hardware
//
#include <hip/hip_runtime.h>
#include <hip/hip_bf16.h>

// ---------------------------------------------------------------------------
// Mamba-2 (SSD) layer forward for MI455X / gfx1250.
// All matmuls via v_wmma_f32_16x16x32_bf16 (wave32 WMMA), f32 accumulate.
// ---------------------------------------------------------------------------

typedef __bf16 bf16_t;
typedef __attribute__((ext_vector_type(16))) __bf16 v16bf;
typedef __attribute__((ext_vector_type(8)))  __bf16 v8bf;
typedef __attribute__((ext_vector_type(8)))  float  v8f;

#define BSZ     8
#define SEQL    2048
#define DMODEL  1024
#define DINNER  2048
#define NHEADS  32
#define HEADDIM 64
#define DSTATE  128
#define CONVDIM 2304
#define DINPROJ 4384
#define CHUNK   128
#define NCHUNK  16
#define MTOT    (BSZ*SEQL)          // 16384 rows

__device__ __forceinline__ v8f wmma_bf(v16bf a, v16bf b, v8f c) {
  // D = A(16x32) * B(32x16) + C(16x16), f32 accumulate
  return __builtin_amdgcn_wmma_f32_16x16x32_bf16(false, a, false, b, (short)0, c,
                                                 false, false);
}

__device__ __forceinline__ v8f zero8() {
  v8f z;
#pragma unroll
  for (int i = 0; i < 8; ++i) z[i] = 0.0f;
  return z;
}

// Build a bf16 A-fragment from two contiguous 8-float chunks (ISA A layout:
// elems 0..7 = K kb..kb+7, elems 8..15 = K kb+16..kb+23).
__device__ __forceinline__ v16bf cvt_pair(v8f lo, v8f hi) {
  v16bf r;
#pragma unroll
  for (int e = 0; e < 8; ++e) { r[e] = (__bf16)lo[e]; r[e + 8] = (__bf16)hi[e]; }
  return r;
}

// ---------------------------------------------------------------------------
// f32 -> bf16 conversion (grid-stride)
// ---------------------------------------------------------------------------
__global__ void k_cvt(const float* __restrict__ in, bf16_t* __restrict__ out, size_t n) {
  size_t i = (size_t)blockIdx.x * blockDim.x + threadIdx.x;
  size_t stride = (size_t)gridDim.x * blockDim.x;
  for (; i < n; i += stride) out[i] = (bf16_t)in[i];
}

// ---------------------------------------------------------------------------
// Generic GEMM: C[M,N] = A[M,K](bf16) @ W[N,K](bf16)^T, f32 out.
// Block tile 64(M) x 128(N), 8 waves, 32x32 per wave, K step 32.
// M % 64 == 0, K % 32 == 0; N guarded (clamped loads, guarded stores).
// ---------------------------------------------------------------------------
__global__ __launch_bounds__(256) void k_gemm_bf16(
    const bf16_t* __restrict__ A, const bf16_t* __restrict__ W,
    float* __restrict__ C, int M, int N, int K) {
  const int lane = threadIdx.x & 31;
  const int wave = threadIdx.x >> 5;
  const int m0 = blockIdx.y * 64 + (wave & 1) * 32;
  const int n0 = blockIdx.x * 128 + (wave >> 1) * 32;
  const int kbA = (lane & 16) ? 8 : 0;
  const int kbW = (lane & 16) ? 8 : 0;

  const bf16_t* pa0 = A + (size_t)(m0 + (lane & 15)) * K + kbA;
  const bf16_t* pa1 = pa0 + (size_t)16 * K;
  int nw0 = n0 + (lane & 15);      if (nw0 > N - 1) nw0 = N - 1;
  int nw1 = n0 + 16 + (lane & 15); if (nw1 > N - 1) nw1 = N - 1;
  const bf16_t* pw0 = W + (size_t)nw0 * K + kbW;
  const bf16_t* pw1 = W + (size_t)nw1 * K + kbW;

  v8f acc00 = zero8(), acc01 = zero8(), acc10 = zero8(), acc11 = zero8();

  for (int k0 = 0; k0 < K; k0 += 32) {
    v16bf a0, a1;
    { union { v16bf v; v8bf h[2]; } u;
      u.h[0] = *(const v8bf*)pa0; u.h[1] = *(const v8bf*)(pa0 + 16); a0 = u.v; }
    { union { v16bf v; v8bf h[2]; } u;
      u.h[0] = *(const v8bf*)pa1; u.h[1] = *(const v8bf*)(pa1 + 16); a1 = u.v; }
    v16bf w0, w1;
    { union { v16bf v; v8bf h[2]; } u;
      u.h[0] = *(const v8bf*)pw0; u.h[1] = *(const v8bf*)(pw0 + 16); w0 = u.v; }
    { union { v16bf v; v8bf h[2]; } u;
      u.h[0] = *(const v8bf*)pw1; u.h[1] = *(const v8bf*)(pw1 + 16); w1 = u.v; }
    __builtin_prefetch(pw0 + 256, 0, 1);
    acc00 = wmma_bf(a0, w0, acc00);
    acc01 = wmma_bf(a0, w1, acc01);
    acc10 = wmma_bf(a1, w0, acc10);
    acc11 = wmma_bf(a1, w1, acc11);
    pa0 += 32; pa1 += 32; pw0 += 32; pw1 += 32;
  }

  const int cn0 = n0 + (lane & 15);
  const int cn1 = cn0 + 16;
  const int rb  = m0 + ((lane & 16) ? 8 : 0);
#pragma unroll
  for (int v = 0; v < 8; ++v) {
    if (cn0 < N) {
      C[(size_t)(rb + v) * N + cn0]      = acc00[v];
      C[(size_t)(rb + 16 + v) * N + cn0] = acc10[v];
    }
    if (cn1 < N) {
      C[(size_t)(rb + v) * N + cn1]      = acc01[v];
      C[(size_t)(rb + 16 + v) * N + cn1] = acc11[v];
    }
  }
}

// ---------------------------------------------------------------------------
// dt = softplus(zx[:, 4352+h] + dt_bias[h])
// ---------------------------------------------------------------------------
__global__ void k_dt(const float* __restrict__ zx, const float* __restrict__ dt_bias,
                     float* __restrict__ dt_sp) {
  int idx = blockIdx.x * 256 + threadIdx.x;           // MTOT*NHEADS
  int m = idx >> 5, h = idx & 31;
  float x = zx[(size_t)m * DINPROJ + (DINNER + CONVDIM) + h] + dt_bias[h];
  dt_sp[idx] = (x > 20.0f) ? x : log1pf(__expf(x));
}

// ---------------------------------------------------------------------------
// acs[b,c,h,s] = cumsum_s( dt_sp * (-exp(A_log[h])) ) within each chunk
// ---------------------------------------------------------------------------
__global__ void k_acs(const float* __restrict__ dt_sp, const float* __restrict__ A_log,
                      float* __restrict__ acs) {
  int idx = blockIdx.x * 256 + threadIdx.x;           // BSZ*NCHUNK*NHEADS = 4096
  int h = idx & 31;
  int c = (idx >> 5) & 15;
  int b = idx >> 9;
  float Ah = -__expf(A_log[h]);
  float acc = 0.0f;
  size_t mrow = (size_t)(b * SEQL + c * CHUNK);
  size_t obase = (size_t)idx * CHUNK;
  for (int s = 0; s < CHUNK; ++s) {
    acc += dt_sp[(mrow + s) * NHEADS + h] * Ah;
    acs[obase + s] = acc;
  }
}

// ---------------------------------------------------------------------------
// Depthwise causal conv (width 4) + bias + SiLU over xBC, with routing:
//   ch < 2048           -> x_f32[m,ch];  xdt_t[b,c,h,p,s] = bf16(x*dt)
//   2048 <= ch < 2176   -> B_bf[m,n] and Bt_bf[b,c,n,s]
//   else                -> C_f32[m,n]
// ---------------------------------------------------------------------------
__global__ void k_conv(const float* __restrict__ zx, const float* __restrict__ conv_w,
                       const float* __restrict__ conv_b, const float* __restrict__ dt_sp,
                       float* __restrict__ x_f32, bf16_t* __restrict__ xdt_t,
                       bf16_t* __restrict__ B_bf, bf16_t* __restrict__ Bt_bf,
                       float* __restrict__ C_f32) {
  size_t idx = (size_t)blockIdx.x * 256 + threadIdx.x;
  if (idx >= (size_t)MTOT * CONVDIM) return;
  int ch = (int)(idx % CONVDIM);
  int m  = (int)(idx / CONVDIM);
  int b = m >> 11, l = m & 2047;
  const float* wr = conv_w + (size_t)ch * 4;
  float acc = conv_b[ch];
#pragma unroll
  for (int j = 0; j < 4; ++j) {
    int ls = l - 3 + j;
    if (ls >= 0) acc += zx[(size_t)(b * SEQL + ls) * DINPROJ + DINNER + ch] * wr[j];
  }
  float v = acc / (1.0f + __expf(-acc));              // SiLU
  int c = l >> 7, s = l & 127;
  if (ch < DINNER) {
    x_f32[(size_t)m * DINNER + ch] = v;
    int h = ch >> 6, p = ch & 63;
    xdt_t[((((size_t)(b * NCHUNK + c)) * NHEADS + h) * HEADDIM + p) * CHUNK + s] =
        (bf16_t)(v * dt_sp[(size_t)m * NHEADS + h]);
  } else if (ch < DINNER + DSTATE) {
    int n = ch - DINNER;
    bf16_t bv = (bf16_t)v;
    B_bf[(size_t)m * DSTATE + n] = bv;
    Bt_bf[(((size_t)(b * NCHUNK + c)) * DSTATE + n) * CHUNK + s] = bv;
  } else {
    C_f32[(size_t)m * DSTATE + (ch - DINNER - DSTATE)] = v;
  }
}

// ---------------------------------------------------------------------------
// CB[bc][l][s] = sum_n C[l,n] * B[s,n]   (128x128, K=128) -- one block per chunk
// 8 waves: wave = (wm 0..3, wn 0..1) -> 32(l) x 64(s) tile each.
// ---------------------------------------------------------------------------
__global__ __launch_bounds__(256) void k_cb(const float* __restrict__ C_f32,
                                            const bf16_t* __restrict__ B_bf,
                                            float* __restrict__ CB) {
  int bc = blockIdx.x;
  int lane = threadIdx.x & 31, wave = threadIdx.x >> 5;
  int l0 = (wave & 3) * 32;
  int s0 = (wave >> 2) * 64;
  int kbA = (lane & 16) ? 8 : 0;
  int kbW = (lane & 16) ? 8 : 0;
  size_t mbase = (size_t)((bc >> 4) * SEQL + (bc & 15) * CHUNK);
  const float* crow0 = C_f32 + (mbase + l0 + (lane & 15)) * DSTATE;
  const float* crow1 = crow0 + (size_t)16 * DSTATE;

  v8f acc[2][4];
#pragma unroll
  for (int i = 0; i < 2; ++i)
#pragma unroll
    for (int t = 0; t < 4; ++t) acc[i][t] = zero8();

  for (int k0 = 0; k0 < DSTATE; k0 += 32) {
    v16bf a0 = cvt_pair(*(const v8f*)(crow0 + k0 + kbA), *(const v8f*)(crow0 + k0 + kbA + 16));
    v16bf a1 = cvt_pair(*(const v8f*)(crow1 + k0 + kbA), *(const v8f*)(crow1 + k0 + kbA + 16));
#pragma unroll
    for (int t = 0; t < 4; ++t) {
      int srow = s0 + t * 16 + (lane & 15);
      const bf16_t* pw = B_bf + (mbase + srow) * DSTATE + k0 + kbW;
      union { v16bf v; v8bf h[2]; } u;
      u.h[0] = *(const v8bf*)pw; u.h[1] = *(const v8bf*)(pw + 16);
      v16bf w = u.v;
      acc[0][t] = wmma_bf(a0, w, acc[0][t]);
      acc[1][t] = wmma_bf(a1, w, acc[1][t]);
    }
  }
  int rm = (lane & 16) ? 8 : 0;
  float* out = CB + (size_t)bc * CHUNK * CHUNK;
#pragma unroll
  for (int i = 0; i < 2; ++i)
#pragma unroll
    for (int t = 0; t < 4; ++t) {
      int scol = s0 + t * 16 + (lane & 15);
#pragma unroll
      for (int v = 0; v < 8; ++v)
        out[(size_t)(l0 + i * 16 + rm + v) * CHUNK + scol] = acc[i][t][v];
    }
}

// ---------------------------------------------------------------------------
// Y_diag[l,p] = sum_s (CB[l,s] * exp(acs_l - acs_s) * [s<=l]) * xdt_t[p,s]
// One block per (bc, h); 8 waves x (16 l-rows, 64 p-cols).  Writes y (=).
// ---------------------------------------------------------------------------
__global__ __launch_bounds__(256) void k_ydiag(const float* __restrict__ CB,
                                               const float* __restrict__ acs,
                                               const bf16_t* __restrict__ xdt_t,
                                               float* __restrict__ y) {
  int bc = blockIdx.x, h = blockIdx.y;
  __shared__ float s_acs[CHUNK];
  if (threadIdx.x < CHUNK)
    s_acs[threadIdx.x] = acs[((size_t)bc * NHEADS + h) * CHUNK + threadIdx.x];
  __syncthreads();

  int lane = threadIdx.x & 31, wave = threadIdx.x >> 5;
  int l0 = wave * 16;
  int lA = l0 + (lane & 15);
  float acs_l = s_acs[lA];
  int kbA = (lane & 16) ? 8 : 0;
  int kbW = (lane & 16) ? 8 : 0;
  const float*  cbrow = CB + ((size_t)bc * CHUNK + lA) * CHUNK;
  const bf16_t* xbase = xdt_t + ((size_t)bc * NHEADS + h) * HEADDIM * CHUNK;

  v8f acc[4];
#pragma unroll
  for (int t = 0; t < 4; ++t) acc[t] = zero8();

  for (int k0 = 0; k0 < CHUNK; k0 += 32) {
    v8f c0 = *(const v8f*)(cbrow + k0 + kbA);
    v8f c1 = *(const v8f*)(cbrow + k0 + kbA + 16);
    v16bf a;
#pragma unroll
    for (int e = 0; e < 8; ++e) {
      int sA = k0 + kbA + e;
      int sB = sA + 16;
      float va = (sA <= lA) ? c0[e] * __expf(acs_l - s_acs[sA]) : 0.0f;
      float vb = (sB <= lA) ? c1[e] * __expf(acs_l - s_acs[sB]) : 0.0f;
      a[e] = (__bf16)va; a[e + 8] = (__bf16)vb;
    }
#pragma unroll
    for (int t = 0; t < 4; ++t) {
      int p = t * 16 + (lane & 15);
      const bf16_t* pw = xbase + (size_t)p * CHUNK + k0 + kbW;
      union { v16bf v; v8bf h[2]; } u;
      u.h[0] = *(const v8bf*)pw; u.h[1] = *(const v8bf*)(pw + 16);
      v16bf w = u.v;
      acc[t] = wmma_bf(a, w, acc[t]);
    }
  }
  int mbase = (bc >> 4) * SEQL + (bc & 15) * CHUNK;
  int rm = (lane & 16) ? 8 : 0;
#pragma unroll
  for (int t = 0; t < 4; ++t) {
    int p = t * 16 + (lane & 15);
#pragma unroll
    for (int v = 0; v < 8; ++v)
      y[((size_t)(mbase + l0 + rm + v) * NHEADS + h) * HEADDIM + p] = acc[t][v];
  }
}

// ---------------------------------------------------------------------------
// states[bc,h,p,n] = sum_s xdt_t[p,s] * exp(acs_last - acs_s) * Bt[n,s]
// One block per (bc, h); 8 waves x (16 p-rows, 64 n-cols).
// ---------------------------------------------------------------------------
__global__ __launch_bounds__(256) void k_states(const float* __restrict__ acs,
                                                const bf16_t* __restrict__ xdt_t,
                                                const bf16_t* __restrict__ Bt,
                                                float* __restrict__ states) {
  int bc = blockIdx.x, h = blockIdx.y;
  __shared__ float s_acs[CHUNK];
  __shared__ float s_dec[CHUNK];
  if (threadIdx.x < CHUNK)
    s_acs[threadIdx.x] = acs[((size_t)bc * NHEADS + h) * CHUNK + threadIdx.x];
  __syncthreads();
  if (threadIdx.x < CHUNK)
    s_dec[threadIdx.x] = __expf(s_acs[CHUNK - 1] - s_acs[threadIdx.x]);
  __syncthreads();

  int lane = threadIdx.x & 31, wave = threadIdx.x >> 5;
  int p0 = (wave & 3) * 16;
  int n0 = (wave >> 2) * 64;
  int kbA = (lane & 16) ? 8 : 0;
  int kbW = (lane & 16) ? 8 : 0;
  const bf16_t* xrow = xdt_t + (((size_t)bc * NHEADS + h) * HEADDIM + p0 + (lane & 15)) * CHUNK;
  const bf16_t* btbase = Bt + (size_t)bc * DSTATE * CHUNK;

  v8f acc[4];
#pragma unroll
  for (int t = 0; t < 4; ++t) acc[t] = zero8();

  for (int k0 = 0; k0 < CHUNK; k0 += 32) {
    v8bf h0 = *(const v8bf*)(xrow + k0 + kbA);
    v8bf h1 = *(const v8bf*)(xrow + k0 + kbA + 16);
    v16bf a;
#pragma unroll
    for (int e = 0; e < 8; ++e) {
      a[e]     = (__bf16)((float)h0[e] * s_dec[k0 + kbA + e]);
      a[e + 8] = (__bf16)((float)h1[e] * s_dec[k0 + kbA + 16 + e]);
    }
#pragma unroll
    for (int t = 0; t < 4; ++t) {
      int n = n0 + t * 16 + (lane & 15);
      const bf16_t* pw = btbase + (size_t)n * CHUNK + k0 + kbW;
      union { v16bf v; v8bf h[2]; } u;
      u.h[0] = *(const v8bf*)pw; u.h[1] = *(const v8bf*)(pw + 16);
      v16bf w = u.v;
      acc[t] = wmma_bf(a, w, acc[t]);
    }
  }
  int rm = (lane & 16) ? 8 : 0;
  float* out = states + ((size_t)bc * NHEADS + h) * HEADDIM * DSTATE;
#pragma unroll
  for (int t = 0; t < 4; ++t) {
    int n = n0 + t * 16 + (lane & 15);
#pragma unroll
    for (int v = 0; v < 8; ++v)
      out[(size_t)(p0 + rm + v) * DSTATE + n] = acc[t][v];
  }
}

// ---------------------------------------------------------------------------
// Inter-chunk scan (in place): states[c] <- carry ; carry = carry*dec_c + st_c
// One thread per (b,h,p,n).
// ---------------------------------------------------------------------------
__global__ void k_scan(float* __restrict__ states, const float* __restrict__ acs) {
  int idx = blockIdx.x * 256 + threadIdx.x;           // 8*32*64*128
  int n = idx & 127;
  int p = (idx >> 7) & 63;
  int h = (idx >> 13) & 31;
  int b = idx >> 18;
  float carry = 0.0f;
  for (int c = 0; c < NCHUNK; ++c) {
    size_t bc = (size_t)(b * NCHUNK + c);
    float dec = __expf(acs[(bc * NHEADS + h) * CHUNK + (CHUNK - 1)]);
    size_t off = ((bc * NHEADS + h) * HEADDIM + p) * DSTATE + n;
    float st = states[off];
    states[off] = carry;
    carry = carry * dec + st;
  }
}

// ---------------------------------------------------------------------------
// Y_off[l,p] += sum_n (C[l,n] * exp(acs_l)) * states_in[p,n]
// Same block/wave shape as k_ydiag; accumulates into y via WMMA C operand.
// ---------------------------------------------------------------------------
__global__ __launch_bounds__(256) void k_yoff(const float* __restrict__ C_f32,
                                              const float* __restrict__ acs,
                                              const float* __restrict__ states,
                                              float* __restrict__ y) {
  int bc = blockIdx.x, h = blockIdx.y;
  __shared__ float s_acs[CHUNK];
  if (threadIdx.x < CHUNK)
    s_acs[threadIdx.x] = acs[((size_t)bc * NHEADS + h) * CHUNK + threadIdx.x];
  __syncthreads();

  int lane = threadIdx.x & 31, wave = threadIdx.x >> 5;
  int l0 = wave * 16;
  int lA = l0 + (lane & 15);
  float scale_l = __expf(s_acs[lA]);
  int kbA = (lane & 16) ? 8 : 0;
  int kbW = (lane & 16) ? 8 : 0;
  int mbase = (bc >> 4) * SEQL + (bc & 15) * CHUNK;
  const float* crow = C_f32 + (size_t)(mbase + lA) * DSTATE;
  const float* sbase = states + ((size_t)bc * NHEADS + h) * HEADDIM * DSTATE;
  int rm = (lane & 16) ? 8 : 0;

  // init accumulators from current y (Y_diag), same layout as the store
  v8f acc[4];
#pragma unroll
  for (int t = 0; t < 4; ++t) {
    int p = t * 16 + (lane & 15);
#pragma unroll
    for (int v = 0; v < 8; ++v)
      acc[t][v] = y[((size_t)(mbase + l0 + rm + v) * NHEADS + h) * HEADDIM + p];
  }

  for (int k0 = 0; k0 < DSTATE; k0 += 32) {
    v8f c0 = *(const v8f*)(crow + k0 + kbA);
    v8f c1 = *(const v8f*)(crow + k0 + kbA + 16);
    v16bf a;
#pragma unroll
    for (int e = 0; e < 8; ++e) {
      a[e]     = (__bf16)(c0[e] * scale_l);
      a[e + 8] = (__bf16)(c1[e] * scale_l);
    }
#pragma unroll
    for (int t = 0; t < 4; ++t) {
      int p = t * 16 + (lane & 15);
      const float* srow = sbase + (size_t)p * DSTATE + k0 + kbW;
      v16bf w = cvt_pair(*(const v8f*)srow, *(const v8f*)(srow + 16));
      acc[t] = wmma_bf(a, w, acc[t]);
    }
  }
#pragma unroll
  for (int t = 0; t < 4; ++t) {
    int p = t * 16 + (lane & 15);
#pragma unroll
    for (int v = 0; v < 8; ++v)
      y[((size_t)(mbase + l0 + rm + v) * NHEADS + h) * HEADDIM + p] = acc[t][v];
  }
}

// ---------------------------------------------------------------------------
// Finalize: y = (y + D[h]*x) * silu(z); RMSNorm over 2048 dims; -> bf16
// One block per row m, 256 threads x 8 elems.
// ---------------------------------------------------------------------------
__global__ __launch_bounds__(256) void k_final(const float* __restrict__ y,
                                               const float* __restrict__ x_f32,
                                               const float* __restrict__ zx,
                                               const float* __restrict__ Dv,
                                               const float* __restrict__ norm_w,
                                               bf16_t* __restrict__ y_bf) {
  int m = blockIdx.x;
  __shared__ float red[256];
  float yg[8];
  float ss = 0.0f;
#pragma unroll
  for (int i = 0; i < 8; ++i) {
    int d = threadIdx.x + i * 256;
    int h = d >> 6;
    float yv = y[(size_t)m * DINNER + d] + Dv[h] * x_f32[(size_t)m * DINNER + d];
    float z = zx[(size_t)m * DINPROJ + d];
    float g = z / (1.0f + __expf(-z));
    yg[i] = yv * g;
    ss += yg[i] * yg[i];
  }
  red[threadIdx.x] = ss;
  __syncthreads();
  for (int off = 128; off > 0; off >>= 1) {
    if (threadIdx.x < off) red[threadIdx.x] += red[threadIdx.x + off];
    __syncthreads();
  }
  float scale = rsqrtf(red[0] / (float)DINNER + 1e-5f);
#pragma unroll
  for (int i = 0; i < 8; ++i) {
    int d = threadIdx.x + i * 256;
    y_bf[(size_t)m * DINNER + d] = (bf16_t)(yg[i] * scale * norm_w[d]);
  }
}

// ---------------------------------------------------------------------------
// pred[m, 0..2] = hidden[m] . dec_w[v] + dec_b[v]; one wave32 per row m.
// ---------------------------------------------------------------------------
__global__ __launch_bounds__(256) void k_dec(const float* __restrict__ hidden,
                                             const float* __restrict__ dec_w,
                                             const float* __restrict__ dec_b,
                                             float* __restrict__ pred) {
  int lane = threadIdx.x & 31, wave = threadIdx.x >> 5;
  int m = blockIdx.x * 8 + wave;
  float a0 = 0.0f, a1 = 0.0f, a2 = 0.0f;
  for (int d = lane; d < DMODEL; d += 32) {
    float hv = hidden[(size_t)m * DMODEL + d];
    a0 += hv * dec_w[d];
    a1 += hv * dec_w[DMODEL + d];
    a2 += hv * dec_w[2 * DMODEL + d];
  }
#pragma unroll
  for (int off = 16; off > 0; off >>= 1) {
    a0 += __shfl_xor(a0, off, 32);
    a1 += __shfl_xor(a1, off, 32);
    a2 += __shfl_xor(a2, off, 32);
  }
  if (lane == 0) {
    pred[(size_t)m * 3 + 0] = a0 + dec_b[0];
    pred[(size_t)m * 3 + 1] = a1 + dec_b[1];
    pred[(size_t)m * 3 + 2] = a2 + dec_b[2];
  }
}

// ---------------------------------------------------------------------------
extern "C" void kernel_launch(void* const* d_in, const int* in_sizes, int n_in,
                              void* d_out, int out_size, void* d_ws, size_t ws_size,
                              hipStream_t stream) {
  (void)in_sizes; (void)n_in; (void)out_size; (void)ws_size;
  const float* embed   = (const float*)d_in[0];
  const float* w_in    = (const float*)d_in[1];
  const float* conv_w  = (const float*)d_in[2];
  const float* conv_b  = (const float*)d_in[3];
  const float* dt_bias = (const float*)d_in[4];
  const float* A_log   = (const float*)d_in[5];
  const float* Dv      = (const float*)d_in[6];
  const float* norm_w  = (const float*)d_in[7];
  const float* w_out   = (const float*)d_in[8];
  const float* dec_w   = (const float*)d_in[9];
  const float* dec_b   = (const float*)d_in[10];

  // ---- workspace layout (~900 MB) ----
  char* ws = (char*)d_ws;
  size_t off = 0;
  auto take = [&](size_t bytes) -> char* {
    char* p = ws + off;
    off = (off + bytes + 255) & ~(size_t)255;
    return p;
  };
  bf16_t* embed_bf = (bf16_t*)take((size_t)MTOT * DMODEL * 2);
  bf16_t* w_in_bf  = (bf16_t*)take((size_t)DINPROJ * DMODEL * 2);
  bf16_t* w_out_bf = (bf16_t*)take((size_t)DMODEL * DINNER * 2);
  float*  zx       = (float*) take((size_t)MTOT * DINPROJ * 4);
  float*  dt_sp    = (float*) take((size_t)MTOT * NHEADS * 4);
  float*  acs      = (float*) take((size_t)BSZ * NCHUNK * NHEADS * CHUNK * 4);
  float*  x_f32    = (float*) take((size_t)MTOT * DINNER * 4);
  bf16_t* xdt_t    = (bf16_t*)take((size_t)BSZ * NCHUNK * NHEADS * HEADDIM * CHUNK * 2);
  bf16_t* B_bf     = (bf16_t*)take((size_t)MTOT * DSTATE * 2);
  bf16_t* Bt_bf    = (bf16_t*)take((size_t)BSZ * NCHUNK * DSTATE * CHUNK * 2);
  float*  C_f32    = (float*) take((size_t)MTOT * DSTATE * 4);
  float*  CBbuf    = (float*) take((size_t)BSZ * NCHUNK * CHUNK * CHUNK * 4);
  float*  states   = (float*) take((size_t)BSZ * NCHUNK * NHEADS * HEADDIM * DSTATE * 4);
  float*  ybuf     = (float*) take((size_t)MTOT * DINNER * 4);
  bf16_t* y_bf     = (bf16_t*)take((size_t)MTOT * DINNER * 2);

  // ---- pipeline ----
  k_cvt<<<4096, 256, 0, stream>>>(embed, embed_bf, (size_t)MTOT * DMODEL);
  k_cvt<<<4096, 256, 0, stream>>>(w_in, w_in_bf, (size_t)DINPROJ * DMODEL);
  k_cvt<<<2048, 256, 0, stream>>>(w_out, w_out_bf, (size_t)DMODEL * DINNER);

  dim3 g1((DINPROJ + 127) / 128, MTOT / 64);
  k_gemm_bf16<<<g1, 256, 0, stream>>>(embed_bf, w_in_bf, zx, MTOT, DINPROJ, DMODEL);

  k_dt<<<(MTOT * NHEADS) / 256, 256, 0, stream>>>(zx, dt_bias, dt_sp);
  k_acs<<<(BSZ * NCHUNK * NHEADS) / 256, 256, 0, stream>>>(dt_sp, A_log, acs);

  size_t convN = (size_t)MTOT * CONVDIM;
  k_conv<<<(unsigned)((convN + 255) / 256), 256, 0, stream>>>(
      zx, conv_w, conv_b, dt_sp, x_f32, xdt_t, B_bf, Bt_bf, C_f32);

  k_cb<<<BSZ * NCHUNK, 256, 0, stream>>>(C_f32, B_bf, CBbuf);

  dim3 gch(BSZ * NCHUNK, NHEADS);
  k_ydiag<<<gch, 256, 0, stream>>>(CBbuf, acs, xdt_t, ybuf);
  k_states<<<gch, 256, 0, stream>>>(acs, xdt_t, Bt_bf, states);
  k_scan<<<(BSZ * NHEADS * HEADDIM * DSTATE) / 256, 256, 0, stream>>>(states, acs);
  k_yoff<<<gch, 256, 0, stream>>>(C_f32, acs, states, ybuf);

  k_final<<<MTOT, 256, 0, stream>>>(ybuf, x_f32, zx, Dv, norm_w, y_bf);

  float* outp   = (float*)d_out;
  float* hidden = outp + (size_t)MTOT * 3;   // (pred | hidden) concat
  dim3 g2(DMODEL / 128, MTOT / 64);
  k_gemm_bf16<<<g2, 256, 0, stream>>>(y_bf, w_out_bf, hidden, MTOT, DMODEL, DINNER);

  k_dec<<<MTOT / 8, 256, 0, stream>>>(hidden, dec_w, dec_b, outp);
}
